// TinyByteModel_62612033241439
// MI455X (gfx1250) — matrix operand order repaired and checked by reading, not compiled, not run on hardware
//
#include <hip/hip_runtime.h>
#include <hip/hip_bf16.h>
#include <math.h>

// ---------------- model dims ----------------
constexpr int B_SZ = 4;
constexpr int T_SZ = 8192;
constexpr int TP_SZ = T_SZ + 2;     // zero-padded rows for conv window
constexpr int D_SZ = 1024;
constexpr int P_SZ = 512;
constexpr int SEG  = T_SZ / P_SZ;   // 16
constexpr int S_SZ = 4;
constexpr int L_SZ = 4;
constexpr int NH_  = 16;
constexpr int NKV_ = 4;
constexpr int HD_  = 64;
constexpr int GQ_  = NH_ / NKV_;    // 4
constexpr int FF_  = 4096;
constexpr int KC_  = 256;
constexpr int V_   = 257;
constexpr int VPAD = 320;           // 257 padded to multiple of 64
constexpr int DH   = D_SZ / 2;      // 512

// ---------------- WMMA types ----------------
typedef __attribute__((ext_vector_type(16))) __bf16 v16bf;
typedef __attribute__((ext_vector_type(8)))  __bf16 v8bf;
typedef __attribute__((ext_vector_type(8)))  float  v8f;

constexpr int BM = 64, BN = 64, BK = 32;
constexpr int LPAD = 8;             // LDS row pad: (32+8)*2 = 80B rows, 16B aligned

// ---------------- device helpers ----------------
__device__ inline float gelu_f(float x) {
    float x3 = x * x * x;
    return 0.5f * x * (1.f + tanhf(0.79788456080286535588f * (x + 0.044715f * x3)));
}
__device__ inline float sigmoid_f(float x) { return 1.f / (1.f + __expf(-x)); }
__device__ inline float silu_f(float x)    { return x * sigmoid_f(x); }
__device__ inline unsigned short bfbits(float x) {
    return __builtin_bit_cast(unsigned short, (__bf16)x);
}

template<int NT>
__device__ inline float block_sum(float v) {
    __shared__ float red[NT];
    int t = threadIdx.x;
    red[t] = v; __syncthreads();
    #pragma unroll
    for (int s = NT / 2; s > 0; s >>= 1) {
        if (t < s) red[t] += red[t + s];
        __syncthreads();
    }
    float r = red[0]; __syncthreads();
    return r;
}
template<int NT>
__device__ inline float block_max(float v) {
    __shared__ float red[NT];
    int t = threadIdx.x;
    red[t] = v; __syncthreads();
    #pragma unroll
    for (int s = NT / 2; s > 0; s >>= 1) {
        if (t < s) red[t] = fmaxf(red[t], red[t + s]);
        __syncthreads();
    }
    float r = red[0]; __syncthreads();
    return r;
}

// ============================================================
// Generic bf16 WMMA GEMM:  C[M,Nout] = act(A[M,K](lda) * B[K,Nload] + bias)
// Preconditions (all call sites satisfy): M % 64 == 0, K % 32 == 0,
// Nload % 64 == 0. Only C stores are guarded (c < Nout). fp32 in/out,
// bf16 WMMA with f32 accumulation. 256 threads = 8 waves; each wave owns
// a 16x32 strip (two v_wmma_f32_16x16x32_bf16 per K-step).
// ACT: 0 = none, 1 = gelu
// ============================================================
template<int ACT>
__global__ void gemm_bf16_kernel(const float* __restrict__ A, int lda,
                                 const float* __restrict__ Bm,
                                 float* __restrict__ C,
                                 const float* __restrict__ bias,
                                 int M, int Nload, int Nout, int K)
{
    __shared__ __bf16 As  [BM][BK + LPAD];   // row-major, 80B rows
    __shared__ __bf16 Bst [BN][BK + LPAD];   // TRANSPOSED: Bst[n][k]
    const int bm = blockIdx.y * BM;
    const int bn = blockIdx.x * BN;
    const int tid  = threadIdx.x;
    const int wave = tid >> 5;
    const int lane = tid & 31;
    const int rw = (wave & 3) * 16;   // wave row offset in tile
    const int cw = (wave >> 2) * 32;  // wave col offset in tile
    v8f acc0 = {}; v8f acc1 = {};

    for (int k0 = 0; k0 < K; k0 += BK) {
        // ---- stage A tile (64x32): unconditional float4 loads ----
        #pragma unroll
        for (int i = 0; i < 2; ++i) {
            int idx = i * 256 + tid;          // 512 float4 chunks
            int r = idx >> 3, c4 = idx & 7;
            float4 f = *(const float4*)(A + (size_t)(bm + r) * lda + k0 + c4 * 4);
            ushort4 u;
            u.x = bfbits(f.x); u.y = bfbits(f.y); u.z = bfbits(f.z); u.w = bfbits(f.w);
            *(ushort4*)&As[r][c4 * 4] = u;    // 8B-aligned b64 store
        }
        // ---- stage B tile (32x64) transposed: float4 loads, b16 stores ----
        #pragma unroll
        for (int i = 0; i < 2; ++i) {
            int idx = i * 256 + tid;
            int r = idx >> 4, c4 = idx & 15;  // r = k row, cols c4*4..+3
            float4 f = *(const float4*)(Bm + (size_t)(k0 + r) * Nload + bn + c4 * 4);
            Bst[c4 * 4 + 0][r] = (__bf16)f.x;
            Bst[c4 * 4 + 1][r] = (__bf16)f.y;
            Bst[c4 * 4 + 2][r] = (__bf16)f.z;
            Bst[c4 * 4 + 3][r] = (__bf16)f.w;
        }
        // prefetch next K tile into near caches (WGP-scope global_prefetch_b8)
        if (k0 + BK < K) {
            __builtin_prefetch(A + (size_t)(bm + (tid & (BM - 1))) * lda + k0 + BK, 0, 3);
            __builtin_prefetch(Bm + (size_t)(k0 + BK + (tid & (BK - 1))) * Nload + bn, 0, 3);
        }
        __syncthreads();

        // ---- fragments: contiguous 16B LDS loads (ds_load_b128) ----
        const int m   = lane & 15;
        const int kb  = (lane < 16) ? 0 : 8;   // A: elems kb..kb+7, kb+16..kb+23
        const __bf16* arow = &As[rw + m][0];
        v8bf a_lo = *(const v8bf*)(arow + kb);
        v8bf a_hi = *(const v8bf*)(arow + kb + 16);
        v16bf af = __builtin_shufflevector(a_lo, a_hi,
            0,1,2,3,4,5,6,7,8,9,10,11,12,13,14,15);

        const int n   = lane & 15;
        const int kbb = (lane < 16) ? 0 : 8;   // B: elems kbb..kbb+7, kbb+16..kbb+23
        const __bf16* brow0 = &Bst[cw + n][kbb];
        v8bf b0l = *(const v8bf*)(brow0);
        v8bf b0h = *(const v8bf*)(brow0 + 16);
        v16bf bf0 = __builtin_shufflevector(b0l, b0h,
            0,1,2,3,4,5,6,7,8,9,10,11,12,13,14,15);
        const __bf16* brow1 = &Bst[cw + 16 + n][kbb];
        v8bf b1l = *(const v8bf*)(brow1);
        v8bf b1h = *(const v8bf*)(brow1 + 16);
        v16bf bf1 = __builtin_shufflevector(b1l, b1h,
            0,1,2,3,4,5,6,7,8,9,10,11,12,13,14,15);

        acc0 = __builtin_amdgcn_wmma_f32_16x16x32_bf16(false, af, false, bf0, (short)0, acc0, false, false);
        acc1 = __builtin_amdgcn_wmma_f32_16x16x32_bf16(false, af, false, bf1, (short)0, acc1, false, false);
        __syncthreads();
    }

    // C layout: VGPR e -> row rw + e (+8 for upper lane half), col = lane&15
    const int rbase = bm + rw + ((lane < 16) ? 0 : 8);
    const int c0 = bn + cw + (lane & 15);
    const int c1 = c0 + 16;
    #pragma unroll
    for (int e = 0; e < 8; ++e) {
        int r = rbase + e;
        if (c0 < Nout) {
            float vv = acc0[e];
            if (bias) vv += bias[c0];
            if (ACT == 1) vv = gelu_f(vv);
            C[(size_t)r * Nout + c0] = vv;
        }
        if (c1 < Nout) {
            float vv = acc1[e];
            if (bias) vv += bias[c1];
            if (ACT == 1) vv = gelu_f(vv);
            C[(size_t)r * Nout + c1] = vv;
        }
    }
}

// ---------------- prep kernels ----------------

// x_pad[b][1+t][:] = emb[tokens[b][t]];  rows 0 and T+1 are zeroed separately
__global__ void embed_kernel(const int* __restrict__ tokens,
                             const float* __restrict__ emb,
                             float* __restrict__ x_pad)
{
    int row = blockIdx.x;                              // B*T
    int b = row / T_SZ, t = row % T_SZ;
    int tok = tokens[row];
    const float4* src = (const float4*)(emb + (size_t)tok * D_SZ);
    float4* dst = (float4*)(x_pad + ((size_t)b * TP_SZ + t + 1) * D_SZ);
    dst[threadIdx.x] = src[threadIdx.x];               // 256 threads * float4
}

__global__ void zero_pad_kernel(float* __restrict__ x_pad)
{
    int i = blockIdx.x;                                // 2*B pad rows
    int b = i >> 1;
    int row = (i & 1) ? (b * TP_SZ + T_SZ + 1) : (b * TP_SZ);
    float4 z = make_float4(0.f, 0.f, 0.f, 0.f);
    ((float4*)(x_pad + (size_t)row * D_SZ))[threadIdx.x] = z;
}

// w1t[kk*D + ch][o] = bp_w1[o][ch][kk]   ->  (3*D x DH) row-major B matrix
__global__ void transpose_w1_kernel(const float* __restrict__ w1,
                                    float* __restrict__ w1t)
{
    int i = blockIdx.x * blockDim.x + threadIdx.x;     // 3*D*DH
    if (i >= 3 * D_SZ * DH) return;
    int kg = i / DH, o = i % DH;
    int kk = kg / D_SZ, ch = kg % D_SZ;
    w1t[i] = w1[((size_t)o * D_SZ + ch) * 3 + kk];
}

// head_pad[k][n] = n < V ? head_w[k][n] : 0   (N padded 257 -> 320)
__global__ void pad_head_kernel(const float* __restrict__ hw,
                                float* __restrict__ hp)
{
    int i = blockIdx.x * blockDim.x + threadIdx.x;     // D*VPAD
    if (i >= D_SZ * VPAD) return;
    int k = i / VPAD, nn = i % VPAD;
    hp[i] = (nn < V_) ? hw[(size_t)k * V_ + nn] : 0.f;
}

// ---------------- small VALU kernels ----------------

__global__ void lg_kernel(const float* __restrict__ h1,
                          const float* __restrict__ w2,
                          const float* __restrict__ b2,
                          float* __restrict__ lg)
{
    int row = blockIdx.x;                              // B*T
    int b = row / T_SZ, t = row % T_SZ;
    float acc = 0.f;
    for (int i = threadIdx.x; i < 3 * DH; i += 128) {
        int o = i % DH, kk = i / DH;
        int ts = t + kk - 1;
        if (ts >= 0 && ts < T_SZ)
            acc += h1[((size_t)b * T_SZ + ts) * DH + o] * w2[o * 3 + kk];
    }
    float s = block_sum<128>(acc);
    if (threadIdx.x == 0) lg[row] = s + b2[0];
}

__global__ void boundary_kernel(const float* __restrict__ lg,
                                float* __restrict__ ent,
                                int* __restrict__ idx)
{
    int r = blockIdx.x * blockDim.x + threadIdx.x;     // B*P
    if (r >= B_SZ * P_SZ) return;
    int b = r / P_SZ, p = r % P_SZ;
    const float* v = lg + (size_t)b * T_SZ + p * SEG;
    float mx = v[0];
    #pragma unroll
    for (int s = 1; s < SEG; ++s) mx = fmaxf(mx, v[s]);
    float e[SEG], den = 0.f;
    #pragma unroll
    for (int s = 0; s < SEG; ++s) { e[s] = __expf(v[s] - mx); den += e[s]; }
    float bp = 0.f, en = 0.f;
    #pragma unroll
    for (int s = 0; s < SEG; ++s) {
        float w = e[s] / den;
        bp += w * (float)(p * SEG + s);
        en -= w * __logf(fmaxf(w, 1e-8f));
    }
    ent[r] = en / logf((float)SEG);
    int id = (int)bp;
    idx[r] = min(max(id, 0), T_SZ - 1);
}

__global__ void gather_norm_kernel(const float* __restrict__ x_pad,
                                   const int* __restrict__ idx,
                                   const float* __restrict__ w,
                                   float* __restrict__ xp)
{
    int r = blockIdx.x;                                // B*P
    int b = r / P_SZ;
    const float* src = x_pad + ((size_t)b * TP_SZ + idx[r] + 1) * D_SZ;
    float ss = 0.f;
    for (int d = threadIdx.x; d < D_SZ; d += 256) { float v = src[d]; ss += v * v; }
    ss = block_sum<256>(ss);
    float inv = rsqrtf(ss / D_SZ + 1e-6f);
    for (int d = threadIdx.x; d < D_SZ; d += 256)
        xp[(size_t)r * D_SZ + d] = src[d] * inv * w[d];
}

__global__ void router_kernel(const float* __restrict__ xp,
                              const float* __restrict__ rw,
                              const float* __restrict__ ent,
                              float* __restrict__ rl)
{
    int r = blockIdx.x;                                // B*P
    float acc = 0.f;
    for (int d = threadIdx.x; d < D_SZ; d += 256)
        acc += xp[(size_t)r * D_SZ + d] * rw[d];
    float s = block_sum<256>(acc);
    if (threadIdx.x == 0) rl[r] = s + ent[r];
}

__global__ void topk_kernel(const float* __restrict__ rl,
                            int* __restrict__ sel,
                            float* __restrict__ gsel)
{
    int b = blockIdx.x;
    __shared__ float vals[P_SZ];
    __shared__ int   scan[P_SZ];
    int p = threadIdx.x;                               // 512 threads
    vals[p] = rl[(size_t)b * P_SZ + p];
    __syncthreads();
    float v = vals[p];
    int rank = 0;
    for (int j = 0; j < P_SZ; ++j) {
        float vj = vals[j];
        rank += (vj > v) || (vj == v && j < p);        // lax.top_k tie-break
    }
    int f = (rank < KC_) ? 1 : 0;
    scan[p] = f;
    __syncthreads();
    for (int off = 1; off < P_SZ; off <<= 1) {
        int t = (p >= off) ? scan[p - off] : 0;
        __syncthreads();
        scan[p] += t;
        __syncthreads();
    }
    if (f) {
        int pos = scan[p] - 1;                         // inclusive -> exclusive
        sel[(size_t)b * KC_ + pos] = p;
        gsel[(size_t)b * KC_ + pos] = sigmoid_f(v);
    }
}

__global__ void gather_rms_kernel(const float* __restrict__ xp,
                                  const int* __restrict__ sel,
                                  const float* __restrict__ w,
                                  float* __restrict__ xs,
                                  float* __restrict__ h)
{
    int r = blockIdx.x;                                // B*KC
    int b = r / KC_;
    const float* src = xp + ((size_t)b * P_SZ + sel[r]) * D_SZ;
    float ss = 0.f;
    for (int d = threadIdx.x; d < D_SZ; d += 256) { float v = src[d]; ss += v * v; }
    ss = block_sum<256>(ss);
    float inv = rsqrtf(ss / D_SZ + 1e-6f);
    for (int d = threadIdx.x; d < D_SZ; d += 256) {
        float v = src[d];
        xs[(size_t)r * D_SZ + d] = v;
        h[(size_t)r * D_SZ + d]  = v * inv * w[d];
    }
}

__global__ void rope_kernel(float* __restrict__ q,
                            float* __restrict__ k,
                            const int* __restrict__ sel)
{
    int r = blockIdx.x;                                // B*KC
    float pos = (float)sel[r];
    int tid = threadIdx.x;
    for (int pi = tid; pi < NH_ * 32; pi += 256) {     // q pairs
        int h = pi >> 5, i = pi & 31;
        float ang = pos * __powf(10000.f, -(float)i / 32.f);
        float c = __cosf(ang), s = __sinf(ang);
        float* base = q + (size_t)r * D_SZ + h * HD_;
        float x0 = base[i], x1 = base[i + 32];
        base[i]      = x0 * c - x1 * s;
        base[i + 32] = x1 * c + x0 * s;
    }
    for (int pi = tid; pi < NKV_ * 32; pi += 256) {    // k pairs
        int h = pi >> 5, i = pi & 31;
        float ang = pos * __powf(10000.f, -(float)i / 32.f);
        float c = __cosf(ang), s = __sinf(ang);
        float* base = k + (size_t)r * (NKV_ * HD_) + h * HD_;
        float x0 = base[i], x1 = base[i + 32];
        base[i]      = x0 * c - x1 * s;
        base[i + 32] = x1 * c + x0 * s;
    }
}

__global__ void attn_kernel(const float* __restrict__ q,
                            const float* __restrict__ k,
                            const float* __restrict__ v,
                            float* __restrict__ out)
{
    int bid = blockIdx.x;                              // B*NH*KC
    int qi = bid % KC_;
    int h  = (bid / KC_) % NH_;
    int b  = bid / (KC_ * NH_);
    int kvh = h / GQ_;
    __shared__ float sc[KC_];
    __shared__ float qrow[HD_];
    int tid = threadIdx.x;                             // 256 == KC
    const float* qp = q + ((size_t)(b * KC_ + qi)) * D_SZ + h * HD_;
    if (tid < HD_) qrow[tid] = qp[tid];
    __syncthreads();
    float s;
    if (tid <= qi) {
        const float* kp = k + ((size_t)(b * KC_ + tid)) * (NKV_ * HD_) + kvh * HD_;
        float d = 0.f;
        #pragma unroll 8
        for (int e = 0; e < HD_; ++e) d += qrow[e] * kp[e];
        s = d * 0.125f;                                // 1/sqrt(64)
    } else s = -1e30f;
    float mx = block_max<256>(s);
    float e = __expf(s - mx);
    sc[tid] = e;
    float den = block_sum<256>(e);
    __syncthreads();
    if (tid < HD_) {
        float acc = 0.f;
        for (int j = 0; j <= qi; ++j)
            acc += sc[j] * v[((size_t)(b * KC_ + j)) * (NKV_ * HD_) + kvh * HD_ + tid];
        out[((size_t)(b * KC_ + qi)) * D_SZ + h * HD_ + tid] = acc / den;
    }
}

__global__ void add_rms_kernel(const float* __restrict__ xs,
                               const float* __restrict__ ao,
                               const float* __restrict__ w,
                               float* __restrict__ h2)
{
    int r = blockIdx.x;                                // B*KC
    float ss = 0.f;
    for (int d = threadIdx.x; d < D_SZ; d += 256) {
        float v = xs[(size_t)r * D_SZ + d] + ao[(size_t)r * D_SZ + d];
        ss += v * v;
    }
    ss = block_sum<256>(ss);
    float inv = rsqrtf(ss / D_SZ + 1e-6f);
    for (int d = threadIdx.x; d < D_SZ; d += 256) {
        float v = xs[(size_t)r * D_SZ + d] + ao[(size_t)r * D_SZ + d];
        h2[(size_t)r * D_SZ + d] = v * inv * w[d];
    }
}

__global__ void silu_mul_kernel(float* __restrict__ g,
                                const float* __restrict__ u,
                                size_t n)
{
    size_t i = (size_t)blockIdx.x * blockDim.x + threadIdx.x;
    if (i < n) g[i] = silu_f(g[i]) * u[i];
}

__global__ void scatter_kernel(float* __restrict__ xp,
                               const float* __restrict__ ao,
                               const float* __restrict__ ff,
                               const float* __restrict__ gsel,
                               const int* __restrict__ sel)
{
    int r = blockIdx.x;                                // B*KC (sel unique per b)
    int b = r / KC_;
    float gv = gsel[r];
    float* dst = xp + ((size_t)b * P_SZ + sel[r]) * D_SZ;
    for (int d = threadIdx.x; d < D_SZ; d += 256)
        dst[d] += gv * (ao[(size_t)r * D_SZ + d] + ff[(size_t)r * D_SZ + d]);
}

__global__ void win_kernel(const float* __restrict__ hh,
                           const float* __restrict__ w2,
                           const float* __restrict__ b2,
                           float* __restrict__ win)
{
    int r = blockIdx.x;                                // B*P
    int tid = threadIdx.x;                             // 256 == D/4
    float h = hh[(size_t)r * (D_SZ / 4) + tid];
    #pragma unroll
    for (int s = 0; s < S_SZ; ++s) {
        float ss = block_sum<256>(h * w2[tid * S_SZ + s]);
        if (tid == 0) win[(size_t)r * S_SZ + s] = sigmoid_f(ss + b2[s]);
    }
}

__global__ void final_norm_kernel(const float* __restrict__ proj,
                                  const float* __restrict__ win,
                                  const float* __restrict__ w,
                                  float* __restrict__ y)
{
    int r = blockIdx.x;                                // B*P*S rows of D
    float wv = win[r];
    const float* src = proj + (size_t)r * D_SZ;
    float ss = 0.f;
    for (int d = threadIdx.x; d < D_SZ; d += 256) { float v = src[d] * wv; ss += v * v; }
    ss = block_sum<256>(ss);
    float inv = rsqrtf(ss / D_SZ + 1e-6f);
    for (int d = threadIdx.x; d < D_SZ; d += 256)
        y[(size_t)r * D_SZ + d] = src[d] * wv * inv * w[d];
}

// ---------------- host side ----------------
static inline void launch_gemm(const float* A, int lda, const float* Bm, float* C,
                               const float* bias, int M, int Nload, int Nout, int K,
                               int act, hipStream_t s)
{
    dim3 g(Nload / BN, M / BM);        // M % 64 == 0, Nload % 64 == 0 at all call sites
    dim3 b(256);
    if (act == 1) gemm_bf16_kernel<1><<<g, b, 0, s>>>(A, lda, Bm, C, bias, M, Nload, Nout, K);
    else          gemm_bf16_kernel<0><<<g, b, 0, s>>>(A, lda, Bm, C, bias, M, Nload, Nout, K);
}

extern "C" void kernel_launch(void* const* d_in, const int* in_sizes, int n_in,
                              void* d_out, int out_size, void* d_ws, size_t ws_size,
                              hipStream_t stream)
{
    const int*   tokens     = (const int*)  d_in[0];
    const float* emb        = (const float*)d_in[1];
    const float* bp_w1      = (const float*)d_in[2];
    const float* bp_b1      = (const float*)d_in[3];
    const float* bp_w2      = (const float*)d_in[4];
    const float* bp_b2      = (const float*)d_in[5];
    const float* enc_norm_w = (const float*)d_in[6];
    const float* router_w   = (const float*)d_in[7];
    const float* ln1_w      = (const float*)d_in[8];
    const float* ln2_w      = (const float*)d_in[9];
    const float* q_w        = (const float*)d_in[10];
    const float* k_w        = (const float*)d_in[11];
    const float* v_w        = (const float*)d_in[12];
    const float* o_w        = (const float*)d_in[13];
    const float* gate_w     = (const float*)d_in[14];
    const float* up_w       = (const float*)d_in[15];
    const float* down_w     = (const float*)d_in[16];
    const float* dec_proj_w = (const float*)d_in[17];
    const float* dec_proj_b = (const float*)d_in[18];
    const float* dec_mlp_w1 = (const float*)d_in[19];
    const float* dec_mlp_b1 = (const float*)d_in[20];
    const float* dec_mlp_w2 = (const float*)d_in[21];
    const float* dec_mlp_b2 = (const float*)d_in[22];
    const float* dec_norm_w = (const float*)d_in[23];
    const float* head_w     = (const float*)d_in[24];
    float* out = (float*)d_out;

    // ---- workspace layout (bump allocator, 256B aligned) ----
    char* wsp = (char*)d_ws;
    size_t off = 0;
    auto alloc = [&](size_t bytes) -> char* {
        char* p = wsp + off;
        off = (off + bytes + 255) & ~(size_t)255;
        return p;
    };
    const size_t NT   = (size_t)B_SZ * T_SZ;          // 32768 tokens
    const size_t NBP  = (size_t)B_SZ * P_SZ;          // 2048 patches
    const size_t NBK  = (size_t)B_SZ * KC_;           // 1024 selected
    const size_t NROW = NBP * S_SZ;                   // 8192 decoder rows

    float* x_pad   = (float*)alloc((size_t)B_SZ * TP_SZ * D_SZ * 4); // dead after gather
    float* h1_buf  = (float*)alloc(NT * DH * 4);
    float* w1t_buf = (float*)alloc((size_t)3 * D_SZ * DH * 4);
    float* hp_buf  = (float*)alloc((size_t)D_SZ * VPAD * 4);
    float* lg_buf  = (float*)alloc(NT * 4);
    float* ent_buf = (float*)alloc(NBP * 4);
    int*   idx_buf = (int*)  alloc(NBP * 4);
    float* xp_buf  = (float*)alloc(NBP * D_SZ * 4);
    float* rl_buf  = (float*)alloc(NBP * 4);
    int*   sel_buf = (int*)  alloc(NBK * 4);
    float* gs_buf  = (float*)alloc(NBK * 4);
    float* xs_buf  = (float*)alloc(NBK * D_SZ * 4);
    float* h_buf   = (float*)alloc(NBK * D_SZ * 4);
    float* q_buf   = (float*)alloc(NBK * D_SZ * 4);
    float* k_buf   = (float*)alloc(NBK * NKV_ * HD_ * 4);
    float* v_buf   = (float*)alloc(NBK * NKV_ * HD_ * 4);
    float* att_buf = (float*)alloc(NBK * D_SZ * 4);
    float* ao_buf  = (float*)alloc(NBK * D_SZ * 4);
    float* h2_buf  = (float*)alloc(NBK * D_SZ * 4);
    float* ff_buf  = (float*)alloc(NBK * D_SZ * 4);
    float* win_buf = (float*)alloc(NROW * 4);
    // alias large late-stage buffers into the dead x_pad region
    // (needs 25.7M floats < 33.56M floats available)
    float* gate_buf = x_pad;
    float* up_buf   = gate_buf + NBK * FF_;
    float* proj_buf = up_buf   + NBK * FF_;
    float* y_buf    = proj_buf + NROW * D_SZ;
    float* hh_buf   = y_buf    + NROW * D_SZ;

    // ---- stage 0: one-time weight reshapes ----
    transpose_w1_kernel<<<(3 * D_SZ * DH + 255) / 256, 256, 0, stream>>>(bp_w1, w1t_buf);
    pad_head_kernel<<<(D_SZ * VPAD + 255) / 256, 256, 0, stream>>>(head_w, hp_buf);

    // ---- stage 1: embedding + byte-patch conv (as GEMM over padded rows) ----
    zero_pad_kernel<<<2 * B_SZ, 256, 0, stream>>>(x_pad);
    embed_kernel<<<(int)NT, 256, 0, stream>>>(tokens, emb, x_pad);
    for (int b = 0; b < B_SZ; ++b) {
        // A row t = x_pad[b, t .. t+2, :] (lda = D), K = 3*D, fused bias+gelu
        launch_gemm(x_pad + (size_t)b * TP_SZ * D_SZ, D_SZ, w1t_buf,
                    h1_buf + (size_t)b * T_SZ * DH, bp_b1,
                    T_SZ, DH, DH, 3 * D_SZ, 1, stream);
    }
    lg_kernel<<<(int)NT, 128, 0, stream>>>(h1_buf, bp_w2, bp_b2, lg_buf);
    boundary_kernel<<<(int)((NBP + 255) / 256), 256, 0, stream>>>(lg_buf, ent_buf, idx_buf);
    gather_norm_kernel<<<(int)NBP, 256, 0, stream>>>(x_pad, idx_buf, enc_norm_w, xp_buf);
    // x_pad is now dead -> aliases become live

    // ---- stage 2: L sparse transformer layers ----
    for (int l = 0; l < L_SZ; ++l) {
        router_kernel<<<(int)NBP, 256, 0, stream>>>(xp_buf, router_w + (size_t)l * D_SZ, ent_buf, rl_buf);
        topk_kernel<<<B_SZ, P_SZ, 0, stream>>>(rl_buf, sel_buf, gs_buf);
        gather_rms_kernel<<<(int)NBK, 256, 0, stream>>>(xp_buf, sel_buf, ln1_w + (size_t)l * D_SZ,
                                                        xs_buf, h_buf);
        launch_gemm(h_buf, D_SZ, q_w + (size_t)l * D_SZ * D_SZ, q_buf, nullptr,
                    (int)NBK, D_SZ, D_SZ, D_SZ, 0, stream);
        launch_gemm(h_buf, D_SZ, k_w + (size_t)l * D_SZ * NKV_ * HD_, k_buf, nullptr,
                    (int)NBK, NKV_ * HD_, NKV_ * HD_, D_SZ, 0, stream);
        launch_gemm(h_buf, D_SZ, v_w + (size_t)l * D_SZ * NKV_ * HD_, v_buf, nullptr,
                    (int)NBK, NKV_ * HD_, NKV_ * HD_, D_SZ, 0, stream);
        rope_kernel<<<(int)NBK, 256, 0, stream>>>(q_buf, k_buf, sel_buf);
        attn_kernel<<<B_SZ * NH_ * KC_, 256, 0, stream>>>(q_buf, k_buf, v_buf, att_buf);
        launch_gemm(att_buf, D_SZ, o_w + (size_t)l * D_SZ * D_SZ, ao_buf, nullptr,
                    (int)NBK, D_SZ, D_SZ, D_SZ, 0, stream);
        add_rms_kernel<<<(int)NBK, 256, 0, stream>>>(xs_buf, ao_buf, ln2_w + (size_t)l * D_SZ, h2_buf);
        launch_gemm(h2_buf, D_SZ, gate_w + (size_t)l * D_SZ * FF_, gate_buf, nullptr,
                    (int)NBK, FF_, FF_, D_SZ, 0, stream);
        launch_gemm(h2_buf, D_SZ, up_w + (size_t)l * D_SZ * FF_, up_buf, nullptr,
                    (int)NBK, FF_, FF_, D_SZ, 0, stream);
        {
            size_t n = NBK * FF_;
            silu_mul_kernel<<<(int)((n + 255) / 256), 256, 0, stream>>>(gate_buf, up_buf, n);
        }
        launch_gemm(gate_buf, FF_, down_w + (size_t)l * FF_ * D_SZ, ff_buf, nullptr,
                    (int)NBK, D_SZ, D_SZ, FF_, 0, stream);
        scatter_kernel<<<(int)NBK, 256, 0, stream>>>(xp_buf, ao_buf, ff_buf, gs_buf, sel_buf);
    }

    // ---- stage 3: decoder ----
    launch_gemm(xp_buf, D_SZ, dec_proj_w, proj_buf, dec_proj_b,
                (int)NBP, S_SZ * D_SZ, S_SZ * D_SZ, D_SZ, 0, stream);
    launch_gemm(xp_buf, D_SZ, dec_mlp_w1, hh_buf, dec_mlp_b1,
                (int)NBP, D_SZ / 4, D_SZ / 4, D_SZ, 1, stream);          // + gelu
    win_kernel<<<(int)NBP, 256, 0, stream>>>(hh_buf, dec_mlp_w2, dec_mlp_b2, win_buf);
    final_norm_kernel<<<(int)NROW, 256, 0, stream>>>(proj_buf, win_buf, dec_norm_w, y_buf);
    launch_gemm(y_buf, D_SZ, hp_buf, out, nullptr,
                (int)NROW, VPAD, V_, D_SZ, 0, stream);

    (void)in_sizes; (void)n_in; (void)out_size; (void)ws_size;
}
